// SBM_4690104287294
// MI455X (gfx1250) — matrix operand order repaired and checked by reading, not compiled, not run on hardware
//
#include <hip/hip_runtime.h>

// ---------------------------------------------------------------- constants
constexpr int Bn = 16, Cn = 128, Hn = 96, Wn = 96;
constexpr int HWn = Hn * Wn;                 // 9216
constexpr int OUTD = 83;                     // 9+25+49
constexpr size_t N_BC   = (size_t)Bn * Cn;   // 2048
constexpr size_t NPLANE = (size_t)Bn * Cn * HWn;
constexpr size_t LOGN   = (size_t)Bn * 3 * HWn;

// workspace layout (bytes)
constexpr size_t F = sizeof(float);
constexpr size_t OFF_ENC = 0;
constexpr size_t OFF_DEC = OFF_ENC + N_BC * F;
constexpr size_t OFF_P3  = OFF_DEC + N_BC * F;
constexpr size_t OFF_P5  = OFF_P3 + N_BC * 9 * F;
constexpr size_t OFF_P7  = OFF_P5 + N_BC * 25 * F;
constexpr size_t OFF_LOG = OFF_P7 + N_BC * 49 * F;
constexpr size_t OFF_WHI = OFF_LOG + LOGN * F;
constexpr size_t OFF_WLO = OFF_WHI + (size_t)Cn * 2 * Cn * 2;   // 128x256 f16
constexpr size_t OFF_BRO = OFF_WLO + (size_t)Cn * 2 * Cn * 2;   // 3 planes follow

// ---------------------------------------------------------------- vec types
typedef __attribute__((ext_vector_type(16))) _Float16 v16h;
typedef __attribute__((ext_vector_type(8)))  _Float16 v8h;
typedef __attribute__((ext_vector_type(8)))  float    v8f;
typedef __attribute__((ext_vector_type(4)))  int      v4i;

__device__ __forceinline__ v16h cat8(v8h a, v8h b) {
  return __builtin_shufflevector(a, b, 0,1,2,3,4,5,6,7,8,9,10,11,12,13,14,15);
}

// ---------------------------------------------------------------- k_mean
// One block reduces one (b,c) plane of x_enc (y=0) or x_dec (y=1).
__global__ __launch_bounds__(256) void k_mean(const float* __restrict__ xe,
                                              const float* __restrict__ xd,
                                              float* __restrict__ enc_vec,
                                              float* __restrict__ dec_vec) {
  const int plane = blockIdx.x;
  const int which = blockIdx.y;
  const float* src = (which ? xd : xe) + (size_t)plane * HWn;
  float s = 0.f;
  for (int i = threadIdx.x; i < HWn; i += 256) s += src[i];
  __shared__ float red[8];
  for (int o = 16; o; o >>= 1) s += __shfl_down(s, o);
  if ((threadIdx.x & 31) == 0) red[threadIdx.x >> 5] = s;
  __syncthreads();
  if (threadIdx.x == 0) {
    float tot = 0.f;
    #pragma unroll
    for (int i = 0; i < 8; ++i) tot += red[i];
    (which ? dec_vec : enc_vec)[plane] = tot * (1.f / HWn);
  }
}

// ---------------------------------------------------------------- k_dynker
// Per batch b: s[o] = dec_vec[b].wp[o]; per channel c: softmax(enc*s) segments.
__global__ __launch_bounds__(128) void k_dynker(const float* __restrict__ enc_vec,
                                                const float* __restrict__ dec_vec,
                                                const float* __restrict__ wp,
                                                float* __restrict__ p3,
                                                float* __restrict__ p5,
                                                float* __restrict__ p7) {
  const int b = blockIdx.x;
  const int t = threadIdx.x;
  __shared__ float sv[OUTD];
  __shared__ float dv[Cn];
  dv[t] = dec_vec[b * Cn + t];
  __syncthreads();
  if (t < OUTD) {
    const float* wrow = wp + t * Cn;
    float acc = 0.f;
    #pragma unroll 8
    for (int j = 0; j < Cn; ++j) acc += dv[j] * wrow[j];
    sv[t] = acc;
  }
  __syncthreads();
  const int c  = t;
  const int bc = b * Cn + c;
  const float ev = enc_vec[bc];
  {
    float mx = -1e30f;
    for (int i = 0; i < 9; ++i) mx = fmaxf(mx, ev * sv[i]);
    float sum = 0.f;
    for (int i = 0; i < 9; ++i) sum += __expf(ev * sv[i] - mx);
    const float inv = 1.f / sum;
    for (int i = 0; i < 9; ++i) p3[(size_t)bc * 9 + i] = __expf(ev * sv[i] - mx) * inv;
  }
  {
    float mx = -1e30f;
    for (int i = 0; i < 25; ++i) mx = fmaxf(mx, ev * sv[9 + i]);
    float sum = 0.f;
    for (int i = 0; i < 25; ++i) sum += __expf(ev * sv[9 + i] - mx);
    const float inv = 1.f / sum;
    for (int i = 0; i < 25; ++i) p5[(size_t)bc * 25 + i] = __expf(ev * sv[9 + i] - mx) * inv;
  }
  {
    float mx = -1e30f;
    for (int i = 0; i < 49; ++i) mx = fmaxf(mx, ev * sv[34 + i]);
    float sum = 0.f;
    for (int i = 0; i < 49; ++i) sum += __expf(ev * sv[34 + i] - mx);
    const float inv = 1.f / sum;
    for (int i = 0; i < 49; ++i) p7[(size_t)bc * 49 + i] = __expf(ev * sv[34 + i] - mx) * inv;
  }
}

// ---------------------------------------------------------------- k_zero
__global__ __launch_bounds__(256) void k_zero(float* __restrict__ p, int n) {
  int i = blockIdx.x * 256 + threadIdx.x;
  if (i < n) p[i] = 0.f;
}

// ---------------------------------------------------------------- k_wsplit
// final_w fp32 -> Markidis hi/lo f16 pair.
__global__ __launch_bounds__(256) void k_wsplit(const float* __restrict__ fw,
                                                _Float16* __restrict__ whi,
                                                _Float16* __restrict__ wlo) {
  int i = blockIdx.x * 256 + threadIdx.x;
  if (i < Cn * 2 * Cn) {
    float v = fw[i];
    _Float16 h = (_Float16)v;
    whi[i] = h;
    wlo[i] = (_Float16)(v - (float)h);
  }
}

// ---------------------------------------------------------------- k_branches
// One block = one 32x32 tile of one (b,c) plane. LDS: x_enc tile + halo4,
// 3 branch tiles + halo1. Produces branch_outs and atomic mask-logit partials.
// The x_enc tile is staged via gfx1250 async global->LDS DMA when available.
__global__ __launch_bounds__(256) void k_branches(const float* __restrict__ xe,
                                                  const float* __restrict__ p3,
                                                  const float* __restrict__ p5,
                                                  const float* __restrict__ p7,
                                                  const float* __restrict__ dw_w,
                                                  const float* __restrict__ pw_w,
                                                  float* __restrict__ bro,
                                                  float* __restrict__ logits) {
  constexpr int TS = 32;
  const int tile = blockIdx.x;           // 0..8
  const int bc   = blockIdx.y;           // 0..2047
  const int b = bc / Cn, c = bc % Cn;
  const int tx0 = (tile % 3) * TS, ty0 = (tile / 3) * TS;
  const int x0 = tx0 - 4, y0 = ty0 - 4;
  const int t = threadIdx.x;

  __shared__ alignas(16) float xs[40][40];
  __shared__ float bo[3][34][34];
  __shared__ float pk[OUTD];
  __shared__ float dwk[3][9];

  if (t < 9)       pk[t] = p3[(size_t)bc * 9 + t];
  else if (t < 34) pk[t] = p5[(size_t)bc * 25 + (t - 9)];
  else if (t < 83) pk[t] = p7[(size_t)bc * 49 + (t - 34)];
  if (t >= 96 && t < 123) {
    int i = t - 96, nb = i / 9, j = i % 9;
    dwk[nb][j] = dw_w[(size_t)(nb * Cn + c) * 9 + j];
  }

  const float* plane = xe + (size_t)bc * HWn;
  if (t < 8) __builtin_prefetch(plane + (size_t)t * 1152, 0, 1);  // global_prefetch_b8

#if defined(__gfx1250__) && __has_builtin(__builtin_amdgcn_global_load_async_to_lds_b128)
  // Zero-fill (covers out-of-range halo), then async-DMA the in-bounds
  // rectangle straight into LDS in 16B chunks (all edges 16B aligned:
  // x starts in {0,28,60}, widths 36/40 floats, LDS row stride 160B).
  {
    float* xsf = &xs[0][0];
    for (int i = t; i < 40 * 40; i += 256) xsf[i] = 0.f;
  }
  __syncthreads();
  {
    const int gx0 = x0 < 0 ? 0 : x0;
    const int gy0 = y0 < 0 ? 0 : y0;
    const int gx1 = (x0 + 40 > Wn) ? Wn : (x0 + 40);
    const int gy1 = (y0 + 40 > Hn) ? Hn : (y0 + 40);
    const int ncx  = (gx1 - gx0) >> 2;    // 16B chunks per row (exact)
    const int nrow = gy1 - gy0;
    for (int i = t; i < nrow * ncx; i += 256) {
      const int r = i / ncx, cx = i % ncx;
      const int gy = gy0 + r, gx = gx0 + cx * 4;
      const float* gp = plane + (size_t)gy * Wn + gx;
      float* lp = &xs[gy - y0][gx - x0];
      __builtin_amdgcn_global_load_async_to_lds_b128(
          (__attribute__((address_space(1))) v4i*)(uintptr_t)gp,
          (__attribute__((address_space(3))) v4i*)(uint32_t)(uintptr_t)lp,
          0, 0);
    }
  }
  #if __has_builtin(__builtin_amdgcn_s_wait_asynccnt)
  __builtin_amdgcn_s_wait_asynccnt(0);
  #else
  asm volatile("s_wait_asynccnt 0" ::: "memory");
  #endif
  __syncthreads();
#else
  for (int i = t; i < 40 * 40; i += 256) {
    int yy = i / 40, xx = i % 40;
    int gy = y0 + yy, gx = x0 + xx;
    xs[yy][xx] = (gy >= 0 && gy < Hn && gx >= 0 && gx < Wn) ? plane[gy * Wn + gx] : 0.f;
  }
  __syncthreads();
#endif

  // branch stencils on 34x34 (halo-1 region around the 32x32 tile)
  for (int i = t; i < 34 * 34; i += 256) {
    int oy = i / 34, ox = i % 34;
    float a3 = 0.f, a5 = 0.f, a7 = 0.f;
    #pragma unroll
    for (int dy = 0; dy < 3; ++dy)
      #pragma unroll
      for (int dx = 0; dx < 3; ++dx)
        a3 += pk[dy * 3 + dx] * xs[oy + 2 + dy][ox + 2 + dx];
    #pragma unroll
    for (int dy = 0; dy < 5; ++dy)
      #pragma unroll
      for (int dx = 0; dx < 5; ++dx)
        a5 += pk[9 + dy * 5 + dx] * xs[oy + 1 + dy][ox + 1 + dx];
    #pragma unroll
    for (int dy = 0; dy < 7; ++dy)
      #pragma unroll
      for (int dx = 0; dx < 7; ++dx)
        a7 += pk[34 + dy * 7 + dx] * xs[oy + dy][ox + dx];
    bo[0][oy][ox] = a3;
    bo[1][oy][ox] = a5;
    bo[2][oy][ox] = a7;
    if (oy >= 1 && oy < 33 && ox >= 1 && ox < 33) {
      int gy = ty0 + oy - 1, gx = tx0 + ox - 1;
      size_t idx = (size_t)bc * HWn + (size_t)gy * Wn + gx;
      bro[idx]              = a3;
      bro[NPLANE + idx]     = a5;
      bro[2 * NPLANE + idx] = a7;
    }
  }
  __syncthreads();

  // 3x3 depthwise conv on branch tiles + partial mask logits (cross-channel sum)
  const float pw00 = pw_w[0 * 384 + 0 * Cn + c], pw01 = pw_w[0 * 384 + Cn + c], pw02 = pw_w[0 * 384 + 2 * Cn + c];
  const float pw10 = pw_w[1 * 384 + 0 * Cn + c], pw11 = pw_w[1 * 384 + Cn + c], pw12 = pw_w[1 * 384 + 2 * Cn + c];
  const float pw20 = pw_w[2 * 384 + 0 * Cn + c], pw21 = pw_w[2 * 384 + Cn + c], pw22 = pw_w[2 * 384 + 2 * Cn + c];
  for (int i = t; i < 32 * 32; i += 256) {
    int oy = i / 32, ox = i % 32;
    float dv0 = 0.f, dv1 = 0.f, dv2 = 0.f;
    #pragma unroll
    for (int dy = 0; dy < 3; ++dy)
      #pragma unroll
      for (int dx = 0; dx < 3; ++dx) {
        dv0 += dwk[0][dy * 3 + dx] * bo[0][oy + dy][ox + dx];
        dv1 += dwk[1][dy * 3 + dx] * bo[1][oy + dy][ox + dx];
        dv2 += dwk[2][dy * 3 + dx] * bo[2][oy + dy][ox + dx];
      }
    int pix = (ty0 + oy) * Wn + (tx0 + ox);
    atomicAdd(&logits[(size_t)(b * 3 + 0) * HWn + pix], dv0 * pw00 + dv1 * pw01 + dv2 * pw02);
    atomicAdd(&logits[(size_t)(b * 3 + 1) * HWn + pix], dv0 * pw10 + dv1 * pw11 + dv2 * pw12);
    atomicAdd(&logits[(size_t)(b * 3 + 2) * HWn + pix], dv0 * pw20 + dv1 * pw21 + dv2 * pw22);
  }
}

// ---------------------------------------------------------------- k_final
// Mask softmax + x_new, then out = final_w @ concat(x_new, x_dec) via
// f32-accumulate f16 WMMA with hi/lo split (3 wmma per K-step of 32).
__global__ __launch_bounds__(256) void k_final(const float* __restrict__ xe,
                                               const float* __restrict__ xd,
                                               const float* __restrict__ bro,
                                               const float* __restrict__ logits,
                                               const float* __restrict__ enc_vec,
                                               const float* __restrict__ lamb_l,
                                               const float* __restrict__ lamb_h,
                                               const float* __restrict__ inside_all,
                                               const _Float16* __restrict__ whi,
                                               const _Float16* __restrict__ wlo,
                                               float* __restrict__ out) {
  constexpr int NP = 32;          // pixels per block (N)
  constexpr int KD = 256;         // contraction depth
  constexpr int BS = 264;         // padded LDS row stride (f16)
  __shared__ alignas(16) _Float16 Bhi[NP][BS];
  __shared__ alignas(16) _Float16 Blo[NP][BS];
  __shared__ float msk[3][NP];

  const int b  = blockIdx.y;
  const int p0 = blockIdx.x * NP;
  const int t  = threadIdx.x;

  if (t < NP) {
    float l0 = logits[(size_t)(b * 3 + 0) * HWn + p0 + t];
    float l1 = logits[(size_t)(b * 3 + 1) * HWn + p0 + t];
    float l2 = logits[(size_t)(b * 3 + 2) * HWn + p0 + t];
    float mx = fmaxf(l0, fmaxf(l1, l2));
    float e0 = __expf(l0 - mx), e1 = __expf(l1 - mx), e2 = __expf(l2 - mx);
    float inv = 1.f / (e0 + e1 + e2);
    msk[0][t] = e0 * inv; msk[1][t] = e1 * inv; msk[2][t] = e2 * inv;
  }
  __syncthreads();

  {
    const int c = t;                         // 0..255 = concat channel (K)
    if (c < Cn) {
      const float ia  = inside_all[c];
      const float ll  = lamb_l[c];
      const float lh  = lamb_h[c] + 1.f;
      const float gap = enc_vec[b * Cn + c];
      const size_t base = (size_t)(b * Cn + c) * HWn + p0;
      const float* q0 = bro + base;
      const float* q1 = bro + NPLANE + base;
      const float* q2 = bro + 2 * NPLANE + base;
      const float* xp = xe + base;
      for (int n = 0; n < NP; ++n) {
        float low = q0[n] * msk[0][n] + q1[n] * msk[1][n] + q2[n] * msk[2][n];
        float v = (low * (ia + 1.f) - ia * gap) * ll + xp[n] * lh;
        _Float16 h = (_Float16)v;
        Bhi[n][c] = h;
        Blo[n][c] = (_Float16)(v - (float)h);
      }
    } else {
      const float* xp = xd + (size_t)(b * Cn + (c - Cn)) * HWn + p0;
      for (int n = 0; n < NP; ++n) {
        float v = xp[n];
        _Float16 h = (_Float16)v;
        Bhi[n][c] = h;
        Blo[n][c] = (_Float16)(v - (float)h);
      }
    }
  }
  __syncthreads();

  const int wave = t >> 5, lane = t & 31;
  const bool hiHalf = lane >= 16;
  const int M0   = wave * 16;
  const int mrow = M0 + (lane & 15);
  const int kb0  = hiHalf ? 16 : 0;    // A fragment K sub-offsets (contiguous: K = 16*half + s)
  const int kb1  = hiHalf ? 24 : 8;

  for (int nt = 0; nt < 2; ++nt) {
    const int n = nt * 16 + (lane & 15);
    v8f acc = {};
    #pragma unroll
    for (int k0 = 0; k0 < KD; k0 += 32) {
      const v8h* ah0 = (const v8h*)(whi + (size_t)mrow * KD + k0 + kb0);
      const v8h* ah1 = (const v8h*)(whi + (size_t)mrow * KD + k0 + kb1);
      const v8h* al0 = (const v8h*)(wlo + (size_t)mrow * KD + k0 + kb0);
      const v8h* al1 = (const v8h*)(wlo + (size_t)mrow * KD + k0 + kb1);
      v16h Ahi = cat8(*ah0, *ah1);
      v16h Alo = cat8(*al0, *al1);
      const int kbb = k0 + (hiHalf ? 16 : 0);   // B: K striped per half-wave
      const v8h* bh = (const v8h*)&Bhi[n][kbb];
      const v8h* bl = (const v8h*)&Blo[n][kbb];
      v16h Bh = cat8(bh[0], bh[1]);
      v16h Bl = cat8(bl[0], bl[1]);
      acc = __builtin_amdgcn_wmma_f32_16x16x32_f16(false, Ahi, false, Bh, (short)0, acc, false, false);
      acc = __builtin_amdgcn_wmma_f32_16x16x32_f16(false, Ahi, false, Bl, (short)0, acc, false, false);
      acc = __builtin_amdgcn_wmma_f32_16x16x32_f16(false, Alo, false, Bh, (short)0, acc, false, false);
    }
    const int mofs = hiHalf ? 8 : 0;            // C/D layout: VGPR i -> M=i / i+8
    const size_t ob = (size_t)b * Cn * HWn + p0 + n;
    #pragma unroll
    for (int i = 0; i < 8; ++i)
      out[ob + (size_t)(M0 + mofs + i) * HWn] = acc[i];
  }
}

// ---------------------------------------------------------------- launch
extern "C" void kernel_launch(void* const* d_in, const int* in_sizes, int n_in,
                              void* d_out, int out_size, void* d_ws, size_t ws_size,
                              hipStream_t stream) {
  (void)in_sizes; (void)n_in; (void)out_size; (void)ws_size;
  const float* x_dec     = (const float*)d_in[0];
  const float* x_enc     = (const float*)d_in[1];
  const float* wp        = (const float*)d_in[2];
  const float* dw_w      = (const float*)d_in[3];
  const float* pw_w      = (const float*)d_in[4];
  const float* lamb_l    = (const float*)d_in[5];
  const float* lamb_h    = (const float*)d_in[6];
  const float* inside_all= (const float*)d_in[7];
  const float* final_w   = (const float*)d_in[8];

  char* ws = (char*)d_ws;
  float*    enc_vec = (float*)(ws + OFF_ENC);
  float*    dec_vec = (float*)(ws + OFF_DEC);
  float*    p3      = (float*)(ws + OFF_P3);
  float*    p5      = (float*)(ws + OFF_P5);
  float*    p7      = (float*)(ws + OFF_P7);
  float*    logits  = (float*)(ws + OFF_LOG);
  _Float16* whi     = (_Float16*)(ws + OFF_WHI);
  _Float16* wlo     = (_Float16*)(ws + OFF_WLO);
  float*    bro     = (float*)(ws + OFF_BRO);

  k_mean  <<<dim3((unsigned)N_BC, 2), 256, 0, stream>>>(x_enc, x_dec, enc_vec, dec_vec);
  k_dynker<<<dim3(Bn), 128, 0, stream>>>(enc_vec, dec_vec, wp, p3, p5, p7);
  k_zero  <<<dim3((unsigned)((LOGN + 255) / 256)), 256, 0, stream>>>(logits, (int)LOGN);
  k_wsplit<<<dim3((Cn * 2 * Cn + 255) / 256), 256, 0, stream>>>(final_w, whi, wlo);
  k_branches<<<dim3(9, (unsigned)N_BC), 256, 0, stream>>>(x_enc, p3, p5, p7, dw_w, pw_w,
                                                          bro, logits);
  k_final <<<dim3(HWn / 32, Bn), 256, 0, stream>>>(x_enc, x_dec, bro, logits, enc_vec,
                                                   lamb_l, lamb_h, inside_all,
                                                   whi, wlo, (float*)d_out);
}
